// AttentionHead_76613626626623
// MI455X (gfx1250) — compile-verified, matrix operand order checked by reading, not run on hardware
//
#include <hip/hip_runtime.h>

// ---------------------------------------------------------------------------
// AttentionHead: dual projection (WMMA bf16) + sigmoid gate + segment-sum
//   x_q = Q @ Wqk^T   [N,64]      x_k = K @ Wk^T  [N,64]
//   e   = sigmoid(sum(x_q*x_k, -1)) * prior
//   out = segment_sum(V * e[:,None], group_idx, G)
// N = 1e6, D = 128, H = 64, G = 50000.  group_idx is sorted.
//
// MI455X plan: HBM floor ~67us (1.57 GB @ 23.3 TB/s). Projections done with
// v_wmma_f32_16x16x32_bf16 (f32 accum). Weights (32 KB bf16) staged in WGP
// LDS once per block so steady-state VMEM is purely the q/k/v HBM stream;
// B fragments come from ds_load_b128 (padded stride -> conflict-free).
// ---------------------------------------------------------------------------

typedef __attribute__((ext_vector_type(16))) __bf16 v16bf;
typedef __attribute__((ext_vector_type(8)))  __bf16 v8bf;
typedef __attribute__((ext_vector_type(8)))  float  v8f;
typedef __attribute__((ext_vector_type(4)))  float  f4;

#define D_DIM 128
#define H_DIM 64
#define TILE_M 16
#define WAVES_PER_BLOCK 8
// Row stride in LDS (bf16 elems): 128 + 8 pad => 272 B/row; adjacent lanes'
// 16 B fragment reads rotate by 4 banks -> conflict-free across a half-wave.
#define LDS_STRIDE 136

// ---- weight conversion: f32 -> bf16 (both matrices into workspace) --------
__global__ void convert_weights_kernel(const float* __restrict__ wq,
                                       const float* __restrict__ wk,
                                       __bf16* __restrict__ dst, int nPer) {
  int i = blockIdx.x * blockDim.x + threadIdx.x;
  if (i < nPer) {
    dst[i] = (__bf16)wq[i];
  } else if (i < 2 * nPer) {
    dst[i] = (__bf16)wk[i - nPer];
  }
}

// ---- zero the output (kernel accumulates via atomics) ---------------------
__global__ void zero_out_kernel(float* __restrict__ out, int n) {
  int i = (blockIdx.x * blockDim.x + threadIdx.x) * 4;
  if (i + 3 < n) {
    f4 z = {};
    *reinterpret_cast<f4*>(out + i) = z;
  } else {
    for (; i < n; ++i) out[i] = 0.f;
  }
}

// ---- fragment loaders -----------------------------------------------------
// 16-bit A-matrix 16x32 layout (ISA 7.12.2): lanes 0-15 row M=lane hold
// K = {kc*32+0..7, +16..23}; lanes 16-31 row M=lane-16 hold {+8..15,+24..31}.
// kbase below already folds in the per-half +8 offset.
__device__ __forceinline__ v16bf load_a_f32(const float* __restrict__ rp,
                                            int kbase) {
  const f4* p0 = reinterpret_cast<const f4*>(rp + kbase);
  const f4* p1 = reinterpret_cast<const f4*>(rp + kbase + 16);
  f4 x0 = p0[0];
  f4 x1 = p0[1];
  f4 x2 = p1[0];
  f4 x3 = p1[1];
  v16bf a;
#pragma unroll
  for (int i = 0; i < 4; ++i) {
    a[i]      = (__bf16)x0[i];
    a[4 + i]  = (__bf16)x1[i];
    a[8 + i]  = (__bf16)x2[i];
    a[12 + i] = (__bf16)x3[i];
  }
  return a;
}

// B = W^T (K=128 x N=64). Column n of B is row n of W (row-major in LDS,
// padded stride). Mirrors the A layout with N in place of M.
__device__ __forceinline__ v16bf load_b_lds(const __bf16* __restrict__ w,
                                            int h, int kbase) {
  const __bf16* r = w + h * LDS_STRIDE + kbase;
  v8bf b0 = *reinterpret_cast<const v8bf*>(r);
  v8bf b1 = *reinterpret_cast<const v8bf*>(r + 16);
  v16bf b;
#pragma unroll
  for (int i = 0; i < 8; ++i) {
    b[i]     = b0[i];
    b[8 + i] = b1[i];
  }
  return b;
}

// ---- main fused kernel: one wave32 per 16-row tile ------------------------
__global__ __launch_bounds__(256) void attn_gate_scatter_kernel(
    const float* __restrict__ q, const float* __restrict__ k,
    const float* __restrict__ v, const int* __restrict__ gidx,
    const float* __restrict__ prior, const __bf16* __restrict__ wq,
    const __bf16* __restrict__ wk, float* __restrict__ out, int nTiles) {
  // Weights live in LDS for the whole block (2 x 64 x 136 bf16 = 34 KB).
  __shared__ __bf16 lds_w[2][H_DIM * LDS_STRIDE];

  // Cooperative fill BEFORE any wave can exit (barrier safety).
  // 2 matrices * 64 rows * 16 chunks of 8 bf16 = 2048 chunks; 8 per thread.
  for (int c = threadIdx.x; c < 2 * H_DIM * (D_DIM / 8); c += blockDim.x) {
    const int mat  = c / (H_DIM * (D_DIM / 8));
    const int rem  = c % (H_DIM * (D_DIM / 8));
    const int row  = rem / (D_DIM / 8);
    const int col  = (rem % (D_DIM / 8)) * 8;
    const __bf16* src = (mat == 0) ? wq : wk;
    v8bf d = *reinterpret_cast<const v8bf*>(src + row * D_DIM + col);
    *reinterpret_cast<v8bf*>(&lds_w[mat][row * LDS_STRIDE + col]) = d;
  }
  __syncthreads();

  const int lane = threadIdx.x & 31;
  const int wave = threadIdx.x >> 5;
  const int tile = blockIdx.x * WAVES_PER_BLOCK + wave;
  if (tile >= nTiles) return;  // wave-uniform: EXEC stays all-1s for WMMA

  const int row0 = tile * TILE_M;
  const int half = lane >> 4;  // which 16-lane half
  const int nl   = lane & 15;

  const float* qrow = q + (size_t)(row0 + nl) * D_DIM;
  const float* krow = k + (size_t)(row0 + nl) * D_DIM;

  // ---- projections: Xq = Qtile @ Wqk^T, Xk = Ktile @ Wk^T (16x64 each) ----
  v8f accq[4] = {};
  v8f acck[4] = {};
#pragma unroll
  for (int kc = 0; kc < 4; ++kc) {       // K = 128 in chunks of 32
    const int kbase = kc * 32 + half * 8;
    v16bf aq = load_a_f32(qrow, kbase);
    v16bf ak = load_a_f32(krow, kbase);
#pragma unroll
    for (int n = 0; n < 4; ++n) {        // H = 64 in tiles of 16
      const int h = n * 16 + nl;
      v16bf bq = load_b_lds(&lds_w[0][0], h, kbase);
      v16bf bk = load_b_lds(&lds_w[1][0], h, kbase);
      accq[n] = __builtin_amdgcn_wmma_f32_16x16x32_bf16(
          false, aq, false, bq, (short)0, accq[n], false, false);
      acck[n] = __builtin_amdgcn_wmma_f32_16x16x32_bf16(
          false, ak, false, bk, (short)0, acck[n], false, false);
    }
  }

  // ---- per-row score: sum_h Xq[m][h]*Xk[m][h] -----------------------------
  // C/D layout: VGPR r, lane l -> (M = r + 8*(l>=16), N = (l&15) + 16n).
  // Reduce the 16 N-columns held across each 16-lane half via xor-shuffles.
  float s[8];
#pragma unroll
  for (int r = 0; r < 8; ++r) {
    float t = 0.f;
#pragma unroll
    for (int n = 0; n < 4; ++n) t += accq[n][r] * acck[n][r];
    t += __shfl_xor(t, 1, 32);
    t += __shfl_xor(t, 2, 32);
    t += __shfl_xor(t, 4, 32);
    t += __shfl_xor(t, 8, 32);
    s[r] = t;  // lanes 0-15: row r ; lanes 16-31: row 8+r
  }

  // ---- gate values + run-compacted scatter (group_idx is sorted) ----------
  const int c0 = lane * 4;  // 128 cols / 32 lanes = 4 per lane
  float a0 = 0.f, a1 = 0.f, a2 = 0.f, a3 = 0.f;
  int cur = gidx[row0];
#pragma unroll
  for (int m = 0; m < TILE_M; ++m) {
    const int g = gidx[row0 + m];
    if (g != cur) {  // uniform across the wave
      float* o = out + (size_t)cur * D_DIM + c0;
      unsafeAtomicAdd(o + 0, a0);
      unsafeAtomicAdd(o + 1, a1);
      unsafeAtomicAdd(o + 2, a2);
      unsafeAtomicAdd(o + 3, a3);
      a0 = a1 = a2 = a3 = 0.f;
      cur = g;
    }
    const float sc   = __shfl(s[m & 7], (m >> 3) << 4, 32);
    const float gate = prior[row0 + m] / (1.f + __expf(-sc));
    const f4 val =
        *reinterpret_cast<const f4*>(v + (size_t)(row0 + m) * D_DIM + c0);
    a0 += val[0] * gate;
    a1 += val[1] * gate;
    a2 += val[2] * gate;
    a3 += val[3] * gate;
  }
  float* o = out + (size_t)cur * D_DIM + c0;
  unsafeAtomicAdd(o + 0, a0);
  unsafeAtomicAdd(o + 1, a1);
  unsafeAtomicAdd(o + 2, a2);
  unsafeAtomicAdd(o + 3, a3);
}

// ---------------------------------------------------------------------------
extern "C" void kernel_launch(void* const* d_in, const int* in_sizes, int n_in,
                              void* d_out, int out_size, void* d_ws,
                              size_t ws_size, hipStream_t stream) {
  const float* q     = (const float*)d_in[0];
  const float* kk    = (const float*)d_in[1];
  const float* v     = (const float*)d_in[2];
  const int*   gidx  = (const int*)d_in[3];
  const float* prior = (const float*)d_in[4];
  const float* wq    = (const float*)d_in[5];
  const float* wk    = (const float*)d_in[6];
  float*       out   = (float*)d_out;

  const int N      = in_sizes[0] / D_DIM;   // 1,000,000
  const int wElems = in_sizes[5];           // 64*128 = 8192

  __bf16* wq_bf = (__bf16*)d_ws;
  __bf16* wk_bf = wq_bf + wElems;

  // 1) weights f32 -> bf16 (32 KB into workspace)
  const int convTotal = 2 * wElems;
  convert_weights_kernel<<<(convTotal + 255) / 256, 256, 0, stream>>>(
      wq, wk, wq_bf, wElems);

  // 2) zero output (we accumulate with atomics; harness replays the graph)
  const int n4 = (out_size + 3) / 4;
  zero_out_kernel<<<(n4 + 255) / 256, 256, 0, stream>>>(out, out_size);

  // 3) fused projection + gate + segment scatter
  const int nTiles = N / TILE_M;  // 62,500 (N divisible by 16)
  const int blocks = (nTiles + WAVES_PER_BLOCK - 1) / WAVES_PER_BLOCK;
  attn_gate_scatter_kernel<<<blocks, 256, 0, stream>>>(
      q, kk, v, gidx, prior, wq_bf, wk_bf, out, nTiles);
}
